// SVGraphConvLayer_35115652612211
// MI455X (gfx1250) — matrix operand order repaired and checked by reading, not compiled, not run on hardware
//
#include <hip/hip_runtime.h>
#include <hip/hip_bf16.h>
#include <math.h>

// GVP graph-conv layer for MI455X (gfx1250, wave32).
//  - prep kernel packs Ws weights to f16 in WMMA B-fragment order (zero-padded K)
//  - edge kernel: gather -> rotate-to-local -> 3 GVP layers (Ws GEMMs on WMMA
//    f32_16x16x32_f16, vector channel on VALU) -> rotate-back -> atomic scatter
//  - node kernel: scatter-mean combine, residual+SV-LN, 2-layer GVP FF in the
//    local frame (same WMMA path), residual+SV-LN, write outputs.
// One wave = 16 rows (edges/nodes); block = 4 waves; K padded to mult. of 32.

typedef __attribute__((ext_vector_type(16))) _Float16 v16h;
typedef __attribute__((ext_vector_type(8)))  _Float16 v8h;
typedef __attribute__((ext_vector_type(8)))  float    v8f;

#define WPB  4      // waves per block
#define ASTR 192    // f16 activation row stride in LDS (max padded K)

#define P0SZ 12288  // packed msg layer0 Ws: 12 ksegs * 64 cols * 16 (f16)
#define PLSZ 6144   // packed 72->64 Ws:      6 ksegs * 64 cols * 16 (f16)

__device__ __forceinline__ float sigmoidf_(float x) { return 1.f / (1.f + __expf(-x)); }

// Pack W (kmax x 64, f32) -> f16 fragment-native order: pack[(kseg*64+col)*16+e]
// holds W[(kseg>>1)*32+(kseg&1)*8+e+(e&8)][col] (0 for K >= kmax). A B-fragment for (chunk ch, tile t,
// lane) is then the 32 contiguous bytes at ((ch*2+sub)*64 + t*16+n)*16.
__global__ void __launch_bounds__(256) pack_weights_kernel(
    const float* Ws0, const float* Ws1, const float* Ws2,
    const float* Ws3, const float* Ws4, _Float16* pack)
{
  const int i = blockIdx.x * blockDim.x + threadIdx.x;
  const int total = P0SZ + 4 * PLSZ;
  if (i >= total) return;
  const float* W;
  int kmax, rem;
  if (i < P0SZ) { W = Ws0; kmax = 162; rem = i; }
  else {
    const int l = (i - P0SZ) / PLSZ;
    rem = (i - P0SZ) % PLSZ;
    W = (l == 0) ? Ws1 : (l == 1) ? Ws2 : (l == 2) ? Ws3 : Ws4;
    kmax = 72;
  }
  const int e = rem & 15, col = (rem >> 4) & 63, kseg = rem >> 10;
  const int K = (kseg >> 1) * 32 + (kseg & 1) * 8 + e + (e & 8);
  pack[i] = (_Float16)((K < kmax) ? W[K * 64 + col] : 0.f);
}

// C(16x64) = relu?(A(16xK) * B(Kx64) + bias). A: f16 rows in LDS (stride ASTR),
// fragments = two contiguous ds_load_b128. B: pre-packed f16, one contiguous
// 32B global read per fragment. Result -> Ssm (f16).
__device__ __forceinline__ void wmma_s_gemm(const _Float16* __restrict__ Asm,
                                            _Float16* __restrict__ Ssm,
                                            const _Float16* __restrict__ Wp,
                                            const float* __restrict__ bias,
                                            int nchunks, bool act, int lane)
{
  const int sub = lane >> 4;        // K-half select (lanes 16-31 hold K 8-15, 24-31)
  const int n   = lane & 15;        // output column within a 16-wide N tile
  const int m   = n;                // A row owned by this lane
  union AB { v16h v; v8h h[2]; };
  v8f acc[4] = {};
  for (int ch = 0; ch < nchunks; ++ch) {
    const _Float16* abase = Asm + m * ASTR + ch * 32 + sub * 8;
    AB a;
    a.h[0] = *(const v8h*)abase;          // fragment elems 0..7  (K base+0..7)
    a.h[1] = *(const v8h*)(abase + 16);   // fragment elems 8..15 (K base+16..23)
    #pragma unroll
    for (int t = 0; t < 4; ++t) {
      const int col = t * 16 + n;
      const v16h b = *(const v16h*)(Wp + (((ch * 2 + sub) * 64) + col) * 16);
      acc[t] = __builtin_amdgcn_wmma_f32_16x16x32_f16(false, a.v, false, b,
                                                      (short)0, acc[t], false, false);
    }
  }
  #pragma unroll
  for (int t = 0; t < 4; ++t) {
    const int col = t * 16 + n;
    const float bv = bias[col];
    #pragma unroll
    for (int r = 0; r < 8; ++r) {   // C: vgpr r -> row r (+8 for upper lanes)
      float v = acc[t][r] + bv;
      if (act) v = fmaxf(v, 0.f);
      Ssm[(r + sub * 8) * 64 + col] = (_Float16)v;
    }
  }
}

// One GVP layer with si=64, vi=h=vo=8 (msg layers 1,2 and both FF layers).
// Inputs: s in Asm[.][0..63] (f16), v in V2 (f32 16x8x3).
// Outputs: s -> Ssm (f16) and copied back to Asm[0..63]; gated v -> V2.
__device__ __forceinline__ void gvp8(_Float16* Asm, _Float16* Ssm,
                                     float* Vh /*16x54*/, float* V2 /*16x24*/,
                                     const float* Wh, const float* Wu,
                                     const _Float16* Wsp, const float* bs,
                                     const float* Wg, const float* bg,
                                     int lane, bool act)
{
  const int m = lane & 15, sub = lane >> 4;
  #pragma unroll
  for (int q = 0; q < 4; ++q) {                 // vh = v @ Wh ; vn -> A[64..71]
    const int h = sub * 4 + q;
    float a0 = 0.f, a1 = 0.f, a2 = 0.f;
    #pragma unroll
    for (int vi = 0; vi < 8; ++vi) {
      const float w = Wh[vi * 8 + h];
      a0 += V2[m * 24 + vi * 3 + 0] * w;
      a1 += V2[m * 24 + vi * 3 + 1] * w;
      a2 += V2[m * 24 + vi * 3 + 2] * w;
    }
    Vh[m * 54 + h * 3 + 0] = a0;
    Vh[m * 54 + h * 3 + 1] = a1;
    Vh[m * 54 + h * 3 + 2] = a2;
    Asm[m * ASTR + 64 + h] = (_Float16)sqrtf(a0 * a0 + a1 * a1 + a2 * a2 + 1e-8f);
  }
  #pragma unroll
  for (int q = 0; q < 12; ++q) Asm[m * ASTR + 72 + sub * 12 + q] = (_Float16)0.f;
  __syncthreads();
  wmma_s_gemm(Asm, Ssm, Wsp, bs, 3, act, lane);
  __syncthreads();
  // gate = sigmoid(s @ Wg + bg); v_out = (vh @ Wu) * gate
  float acc[4];
  #pragma unroll
  for (int q = 0; q < 4; ++q) acc[q] = bg[sub * 4 + q];
  for (int cb = 0; cb < 8; ++cb) {
    const v8h sv = *(const v8h*)&Ssm[m * 64 + cb * 8];
    #pragma unroll
    for (int j = 0; j < 8; ++j) {
      const float s = (float)sv[j];
      const float4 wg = *(const float4*)&Wg[(cb * 8 + j) * 8 + sub * 4];
      acc[0] += s * wg.x; acc[1] += s * wg.y; acc[2] += s * wg.z; acc[3] += s * wg.w;
    }
  }
  #pragma unroll
  for (int q = 0; q < 4; ++q) {
    const int o = sub * 4 + q;
    const float gate = sigmoidf_(acc[q]);
    #pragma unroll
    for (int d = 0; d < 3; ++d) {
      float vu = 0.f;
      #pragma unroll
      for (int h = 0; h < 8; ++h) vu += Vh[m * 54 + h * 3 + d] * Wu[h * 8 + o];
      V2[m * 24 + o * 3 + d] = vu * gate;
    }
  }
  #pragma unroll
  for (int q = 0; q < 4; ++q)                   // s feeds the next layer
    *(v8h*)&Asm[m * ASTR + sub * 32 + q * 8] = *(const v8h*)&Ssm[m * 64 + sub * 32 + q * 8];
  __syncthreads();
}

__global__ void __launch_bounds__(256) zero_ws_kernel(float* p, int n) {
  int i = blockIdx.x * blockDim.x + threadIdx.x;
  if (i < n) p[i] = 0.f;
}

__global__ void __launch_bounds__(128) edge_msg_kernel(
    const float* __restrict__ xs, const float* __restrict__ xv,
    const int* __restrict__ eidx, const float* __restrict__ eas,
    const float* __restrict__ eav, const float* __restrict__ rot,
    const float* Wh0, const float* Wu0, const float* bs0,
    const float* Wg0, const float* bg0,
    const float* Wh1, const float* Wu1, const float* bs1,
    const float* Wg1, const float* bg1,
    const float* Wh2, const float* Wu2, const float* bs2,
    const float* Wg2, const float* bg2,
    const _Float16* __restrict__ pack,
    float* agg_s, float* agg_v, float* cntp, int E)
{
  __shared__ __align__(32) _Float16 Asm[WPB][16][ASTR];  // f16 GEMM activations
  __shared__ __align__(32) _Float16 Ssm[WPB][16][64];    // f16 GEMM outputs
  __shared__ float Vsm[WPB][16][54];                     // mv / vh (18x3)
  __shared__ float V2s[WPB][16][24];                     // gated vectors (8x3)
  __shared__ float Rsm[WPB][16][9];                      // per-edge rotation R_i
  __shared__ int   Ism[WPB][16], Jsm[WPB][16];           // edge endpoints

  const int wv = threadIdx.x >> 5, lane = threadIdx.x & 31;
  const int e0 = (blockIdx.x * WPB + wv) * 16;

  if (lane < 16) {
    const int e = (e0 + lane < E) ? e0 + lane : E - 1;
    Ism[wv][lane] = eidx[e];
    Jsm[wv][lane] = eidx[E + e];
  }
  __syncthreads();

  // ---- gather: ms = [x_s[i], x_s[j], ea_s] (f16), mv = [x_v[i], x_v[j], ea_v]
  for (int t = lane; t < 16 * 144; t += 32) {
    const int me = t / 144, c = t - me * 144;
    const int e = (e0 + me < E) ? e0 + me : E - 1;
    const float v = (c < 64)  ? xs[Ism[wv][me] * 64 + c]
                  : (c < 128) ? xs[Jsm[wv][me] * 64 + (c - 64)]
                              : eas[e * 16 + (c - 128)];
    Asm[wv][me][c] = (_Float16)v;
  }
  for (int t = lane; t < 16 * 54; t += 32) {
    const int me = t / 54, c = t - me * 54;
    const int e = (e0 + me < E) ? e0 + me : E - 1;
    const int vi = c / 3, d = c - vi * 3;
    const float v = (vi < 8)  ? xv[Ism[wv][me] * 24 + vi * 3 + d]
                  : (vi < 16) ? xv[Jsm[wv][me] * 24 + (vi - 8) * 3 + d]
                              : eav[e * 6 + (vi - 16) * 3 + d];
    Vsm[wv][me][c] = v;
  }
  for (int t = lane; t < 16 * 9; t += 32) {
    const int me = t / 9, c = t - me * 9;
    Rsm[wv][me][c] = rot[Ism[wv][me] * 9 + c];
  }
  __syncthreads();

  // ---- rotate mv to local frame: v_loc[k] = sum_d v[d] * R[d][k]
  for (int t = lane; t < 16 * 18; t += 32) {
    const int me = t / 18, vi = t - me * 18;
    const float* R = &Rsm[wv][me][0];
    const float x = Vsm[wv][me][vi * 3 + 0];
    const float y = Vsm[wv][me][vi * 3 + 1];
    const float z = Vsm[wv][me][vi * 3 + 2];
    Vsm[wv][me][vi * 3 + 0] = x * R[0] + y * R[3] + z * R[6];
    Vsm[wv][me][vi * 3 + 1] = x * R[1] + y * R[4] + z * R[7];
    Vsm[wv][me][vi * 3 + 2] = x * R[2] + y * R[5] + z * R[8];
  }
  __syncthreads();

  const int m = lane & 15, sub = lane >> 4;

  // ---- GVP msg layer 0: vi=18, h=18, si=144, K=162 (pad 192)
  float vh[9][3];
  #pragma unroll
  for (int q = 0; q < 9; ++q) {
    const int h = sub * 9 + q;
    float a0 = 0.f, a1 = 0.f, a2 = 0.f;
    for (int vi = 0; vi < 18; ++vi) {
      const float w = Wh0[vi * 18 + h];
      a0 += Vsm[wv][m][vi * 3 + 0] * w;
      a1 += Vsm[wv][m][vi * 3 + 1] * w;
      a2 += Vsm[wv][m][vi * 3 + 2] * w;
    }
    vh[q][0] = a0; vh[q][1] = a1; vh[q][2] = a2;
  }
  __syncthreads();                              // all lanes done reading mv
  #pragma unroll
  for (int q = 0; q < 9; ++q) {
    const int h = sub * 9 + q;
    Vsm[wv][m][h * 3 + 0] = vh[q][0];
    Vsm[wv][m][h * 3 + 1] = vh[q][1];
    Vsm[wv][m][h * 3 + 2] = vh[q][2];
    Asm[wv][m][144 + h] =
        (_Float16)sqrtf(vh[q][0]*vh[q][0] + vh[q][1]*vh[q][1] + vh[q][2]*vh[q][2] + 1e-8f);
  }
  #pragma unroll
  for (int q = 0; q < 15; ++q) Asm[wv][m][162 + sub * 15 + q] = (_Float16)0.f;
  __syncthreads();
  wmma_s_gemm(&Asm[wv][0][0], &Ssm[wv][0][0], pack, bs0, 6, true, lane);
  __syncthreads();
  {                                             // layer-0 gate + vu (h=18)
    float acc[4];
    #pragma unroll
    for (int q = 0; q < 4; ++q) acc[q] = bg0[sub * 4 + q];
    for (int cb = 0; cb < 8; ++cb) {
      const v8h sv = *(const v8h*)&Ssm[wv][m][cb * 8];
      #pragma unroll
      for (int j = 0; j < 8; ++j) {
        const float s = (float)sv[j];
        const float4 wg = *(const float4*)&Wg0[(cb * 8 + j) * 8 + sub * 4];
        acc[0] += s * wg.x; acc[1] += s * wg.y; acc[2] += s * wg.z; acc[3] += s * wg.w;
      }
    }
    #pragma unroll
    for (int q = 0; q < 4; ++q) {
      const int o = sub * 4 + q;
      const float gate = sigmoidf_(acc[q]);
      #pragma unroll
      for (int d = 0; d < 3; ++d) {
        float vu = 0.f;
        for (int h = 0; h < 18; ++h) vu += Vsm[wv][m][h * 3 + d] * Wu0[h * 8 + o];
        V2s[wv][m][o * 3 + d] = vu * gate;
      }
    }
  }
  #pragma unroll
  for (int q = 0; q < 4; ++q)
    *(v8h*)&Asm[wv][m][sub * 32 + q * 8] = *(const v8h*)&Ssm[wv][m][sub * 32 + q * 8];
  __syncthreads();

  // ---- GVP msg layers 1 (relu) and 2 (linear)
  gvp8(&Asm[wv][0][0], &Ssm[wv][0][0], &Vsm[wv][0][0], &V2s[wv][0][0],
       Wh1, Wu1, pack + P0SZ, bs1, Wg1, bg1, lane, true);
  gvp8(&Asm[wv][0][0], &Ssm[wv][0][0], &Vsm[wv][0][0], &V2s[wv][0][0],
       Wh2, Wu2, pack + P0SZ + PLSZ, bs2, Wg2, bg2, lane, false);

  // ---- rotate back + atomic scatter-add (mean divisor accumulated separately)
  const int e = e0 + m;
  const bool valid = e < E;
  const int ii = Ism[wv][m];
  const float* R = &Rsm[wv][m][0];
  #pragma unroll
  for (int q = 0; q < 4; ++q) {
    const int o = sub * 4 + q;
    const float x = V2s[wv][m][o * 3 + 0];
    const float y = V2s[wv][m][o * 3 + 1];
    const float z = V2s[wv][m][o * 3 + 2];
    const float g0 = x * R[0] + y * R[1] + z * R[2];   // v_glob[d]=sum_k v_loc[k]R[d][k]
    const float g1 = x * R[3] + y * R[4] + z * R[5];
    const float g2 = x * R[6] + y * R[7] + z * R[8];
    if (valid) {
      atomicAdd(&agg_v[ii * 24 + o * 3 + 0], g0);
      atomicAdd(&agg_v[ii * 24 + o * 3 + 1], g1);
      atomicAdd(&agg_v[ii * 24 + o * 3 + 2], g2);
    }
  }
  if (valid) {
    for (int q = 0; q < 32; ++q)
      atomicAdd(&agg_s[ii * 64 + sub * 32 + q], (float)Ssm[wv][m][sub * 32 + q]);
    if (sub == 0) atomicAdd(&cntp[ii], 1.0f);
  }
}

__global__ void __launch_bounds__(128) node_update_kernel(
    const float* __restrict__ xs, const float* __restrict__ xv,
    const float* __restrict__ rot,
    const float* fWh0, const float* fWu0, const float* fbs0,
    const float* fWg0, const float* fbg0,
    const float* fWh1, const float* fWu1, const float* fbs1,
    const float* fWg1, const float* fbg1,
    const float* ln1g, const float* ln1b, const float* ln2g, const float* ln2b,
    const _Float16* __restrict__ pack,
    const float* __restrict__ agg_s, const float* __restrict__ agg_v,
    const float* __restrict__ cntp,
    float* out_s, float* out_v, int N)
{
  __shared__ __align__(32) _Float16 Asm[WPB][16][ASTR];
  __shared__ __align__(32) _Float16 Ssm[WPB][16][64];
  __shared__ float Vsm[WPB][16][54];
  __shared__ float V2s[WPB][16][24];
  __shared__ float Rsm[WPB][16][9];

  const int wv = threadIdx.x >> 5, lane = threadIdx.x & 31;
  const int n0 = (blockIdx.x * WPB + wv) * 16;
  const int m = lane & 15, sub = lane >> 4;
  const int nn = (n0 + m < N) ? n0 + m : N - 1;
  const bool valid = (n0 + m) < N;

  const float inv = 1.f / fmaxf(cntp[nn], 1.f);

  // ---- s = LN1(x_s + agg_s/den); keep f32 copy in regs for the residual
  float sreg[32];
  float ps = 0.f;
  for (int q = 0; q < 32; ++q) {
    const int c = sub * 32 + q;
    const float v = xs[nn * 64 + c] + agg_s[nn * 64 + c] * inv;
    sreg[q] = v; ps += v;
  }
  ps += __shfl_xor(ps, 16, 32);
  const float mu = ps * (1.f / 64.f);
  float pv = 0.f;
  for (int q = 0; q < 32; ++q) { const float d = sreg[q] - mu; pv += d * d; }
  pv += __shfl_xor(pv, 16, 32);
  const float rstd = rsqrtf(pv * (1.f / 64.f) + 1e-5f);

  // ---- v = (x_v + agg_v/den) / sqrt(mean ||v||^2 + eps)
  float vreg[12];
  float pn = 0.f;
  #pragma unroll
  for (int q = 0; q < 4; ++q) {
    const int o = sub * 4 + q;
    #pragma unroll
    for (int d = 0; d < 3; ++d) {
      const float v = xv[nn * 24 + o * 3 + d] + agg_v[nn * 24 + o * 3 + d] * inv;
      vreg[q * 3 + d] = v; pn += v * v;
    }
  }
  pn += __shfl_xor(pn, 16, 32);
  const float vni = rsqrtf(pn * (1.f / 8.f) + 1e-8f);
  #pragma unroll
  for (int i = 0; i < 12; ++i) vreg[i] *= vni;

  for (int q = 0; q < 32; ++q) {                // apply LN1 to s, stage as f16
    const int c = sub * 32 + q;
    const float v = (sreg[q] - mu) * rstd * ln1g[c] + ln1b[c];
    sreg[q] = v;
    Asm[wv][m][c] = (_Float16)v;
  }
  for (int t = lane; t < 16 * 9; t += 32) {
    const int me = t / 9, c = t - me * 9;
    const int nb = (n0 + me < N) ? n0 + me : N - 1;
    Rsm[wv][me][c] = rot[nb * 9 + c];
  }
  __syncthreads();

  const float* R = &Rsm[wv][m][0];
  #pragma unroll
  for (int q = 0; q < 4; ++q) {                 // rotate v to local frame
    const int o = sub * 4 + q;
    const float x = vreg[q * 3 + 0], y = vreg[q * 3 + 1], z = vreg[q * 3 + 2];
    V2s[wv][m][o * 3 + 0] = x * R[0] + y * R[3] + z * R[6];
    V2s[wv][m][o * 3 + 1] = x * R[1] + y * R[4] + z * R[7];
    V2s[wv][m][o * 3 + 2] = x * R[2] + y * R[5] + z * R[8];
  }
  __syncthreads();

  gvp8(&Asm[wv][0][0], &Ssm[wv][0][0], &Vsm[wv][0][0], &V2s[wv][0][0],
       fWh0, fWu0, pack + P0SZ + 2 * PLSZ, fbs0, fWg0, fbg0, lane, true);
  gvp8(&Asm[wv][0][0], &Ssm[wv][0][0], &Vsm[wv][0][0], &V2s[wv][0][0],
       fWh1, fWu1, pack + P0SZ + 3 * PLSZ, fbs1, fWg1, fbg1, lane, false);

  // ---- residual + LN2, write outputs
  float s2[32];
  float ps2 = 0.f;
  for (int q = 0; q < 32; ++q) {
    const int c = sub * 32 + q;
    const float v = sreg[q] + (float)Ssm[wv][m][c];
    s2[q] = v; ps2 += v;
  }
  ps2 += __shfl_xor(ps2, 16, 32);
  const float mu2 = ps2 * (1.f / 64.f);
  float pv2 = 0.f;
  for (int q = 0; q < 32; ++q) { const float d = s2[q] - mu2; pv2 += d * d; }
  pv2 += __shfl_xor(pv2, 16, 32);
  const float rstd2 = rsqrtf(pv2 * (1.f / 64.f) + 1e-5f);

  float v2[12];
  float pn2 = 0.f;
  #pragma unroll
  for (int q = 0; q < 4; ++q) {                 // fv back to global frame + residual
    const int o = sub * 4 + q;
    const float x = V2s[wv][m][o * 3 + 0];
    const float y = V2s[wv][m][o * 3 + 1];
    const float z = V2s[wv][m][o * 3 + 2];
    #pragma unroll
    for (int d = 0; d < 3; ++d) {
      const float g = x * R[d * 3 + 0] + y * R[d * 3 + 1] + z * R[d * 3 + 2];
      const float v = vreg[q * 3 + d] + g;
      v2[q * 3 + d] = v; pn2 += v * v;
    }
  }
  pn2 += __shfl_xor(pn2, 16, 32);
  const float vni2 = rsqrtf(pn2 * (1.f / 8.f) + 1e-8f);

  if (valid) {
    for (int q = 0; q < 32; ++q) {
      const int c = sub * 32 + q;
      out_s[nn * 64 + c] = (s2[q] - mu2) * rstd2 * ln2g[c] + ln2b[c];
    }
    #pragma unroll
    for (int q = 0; q < 4; ++q) {
      const int o = sub * 4 + q;
      #pragma unroll
      for (int d = 0; d < 3; ++d)
        out_v[nn * 24 + o * 3 + d] = v2[q * 3 + d] * vni2;
    }
  }
}

extern "C" void kernel_launch(void* const* d_in, const int* in_sizes, int n_in,
                              void* d_out, int out_size, void* d_ws, size_t ws_size,
                              hipStream_t stream) {
  const float* xs  = (const float*)d_in[0];
  const float* xv  = (const float*)d_in[1];
  const int*   ei  = (const int*)  d_in[2];
  const float* eas = (const float*)d_in[3];
  const float* eav = (const float*)d_in[4];
  const float* rot = (const float*)d_in[5];
  // params flattened in dict-insertion order:
  // msg[0..2]{Wh,Wu,Ws,bs,Wg,bg}, ff[0..1]{Wh,Wu,Ws,bs,Wg,bg}, ln1_g, ln1_b, ln2_g, ln2_b
  #define PRM(i) ((const float*)d_in[6 + (i)])
  const int N = in_sizes[0] / 64;
  const int E = in_sizes[2] / 2;

  float* agg_s = (float*)d_ws;
  float* agg_v = agg_s + (size_t)N * 64;
  float* cntp  = agg_v + (size_t)N * 24;
  const size_t aggTot = (((size_t)N * 89) + 7) & ~(size_t)7;  // 32B-align pack area
  _Float16* pack = (_Float16*)((float*)d_ws + aggTot);

  const int zn = N * (64 + 24 + 1);
  zero_ws_kernel<<<(zn + 255) / 256, 256, 0, stream>>>(agg_s, zn);

  const int pn = P0SZ + 4 * PLSZ;
  pack_weights_kernel<<<(pn + 255) / 256, 256, 0, stream>>>(
      PRM(2), PRM(8), PRM(14), PRM(20), PRM(26), pack);  // Ws of msg0..2, ff0..1

  edge_msg_kernel<<<(E + WPB * 16 - 1) / (WPB * 16), WPB * 32, 0, stream>>>(
      xs, xv, ei, eas, eav, rot,
      PRM(0),  PRM(1),  PRM(3),  PRM(4),  PRM(5),    // msg[0] Wh,Wu,bs,Wg,bg
      PRM(6),  PRM(7),  PRM(9),  PRM(10), PRM(11),   // msg[1]
      PRM(12), PRM(13), PRM(15), PRM(16), PRM(17),   // msg[2]
      pack, agg_s, agg_v, cntp, E);

  node_update_kernel<<<(N + WPB * 16 - 1) / (WPB * 16), WPB * 32, 0, stream>>>(
      xs, xv, rot,
      PRM(18), PRM(19), PRM(21), PRM(22), PRM(23),   // ff[0]
      PRM(24), PRM(25), PRM(27), PRM(28), PRM(29),   // ff[1]
      PRM(30), PRM(31), PRM(32), PRM(33),            // ln params
      pack, agg_s, agg_v, cntp,
      (float*)d_out, (float*)d_out + (size_t)N * 64, N);
  #undef PRM
}
